// HybridGnnTorsoV2_74036646248567
// MI455X (gfx1250) — compile-verified, matrix operand order checked by reading, not run on hardware
//
#include <hip/hip_runtime.h>

#define BDIM 256
#define NB   512   // batch
#define TT   16
#define NN   512   // S^3
#define CD   128
#define LGNN 3
#define LTF  2
#define SEQ  15
#define FFD  512

typedef __attribute__((ext_vector_type(16))) __bf16 v16bf;
typedef __attribute__((ext_vector_type(8)))  float  v8f;

union FragBF { v16bf v; unsigned int u[8]; };

__device__ __forceinline__ unsigned short f2bf(float f) {
  union { float f; unsigned int u; } c; c.f = f;
  unsigned int r = c.u + 0x7FFFu + ((c.u >> 16) & 1u);  // round-to-nearest-even
  return (unsigned short)(r >> 16);
}
__device__ __forceinline__ float bf2f(unsigned short h) {
  union { float f; unsigned int u; } c; c.u = ((unsigned int)h) << 16;
  return c.f;
}
__device__ __forceinline__ unsigned int pk2(float lo, float hi) {
  return (unsigned int)f2bf(lo) | ((unsigned int)f2bf(hi) << 16);
}

// ---- CDNA5 async global->LDS copy (ASYNCcnt-tracked, bypasses VGPRs) ----
// GLOBAL_LOAD_ASYNC_TO_LDS_B128, GV mode: per-lane 64-bit global address,
// per-lane 32-bit LDS byte offset (low 32 bits of the flat shared pointer).
__device__ __forceinline__ void async_copy16(unsigned lds_off, const void* gptr) {
  unsigned long long ga = (unsigned long long)gptr;
  asm volatile("global_load_async_to_lds_b128 %0, %1, off"
               :: "v"(lds_off), "v"(ga)
               : "memory");
}
__device__ __forceinline__ void wait_async0() {
  asm volatile("s_wait_asynccnt 0" ::: "memory");
}

// A/B bf16 fragment from LDS (rows are K-contiguous). Layout per ISA 7.12.2:
// lane holds row (lane&15); element i -> K = (i>>3)*16 + (lane>>4)*8 + (i&7),
// i.e. two contiguous 16-byte runs at k0 and k0+16.
__device__ __forceinline__ v16bf load_frag_lds(const unsigned short* rowbase, int kc, int lane) {
  int k0 = kc * 32 + ((lane >> 4) << 3);
  const uint4* p = (const uint4*)(rowbase + k0);
  const uint4* q = (const uint4*)(rowbase + k0 + 16);
  uint4 a = *p, b = *q;
  FragBF f;
  f.u[0] = a.x; f.u[1] = a.y; f.u[2] = a.z; f.u[3] = a.w;
  f.u[4] = b.x; f.u[5] = b.y; f.u[6] = b.z; f.u[7] = b.w;
  return f.v;
}

// B fragment straight from global fp32 weights (rows K-contiguous), cvt->bf16.
__device__ __forceinline__ v16bf load_frag_gf32(const float* rowbase, int kc, int lane) {
  int k0 = kc * 32 + ((lane >> 4) << 3);
  const float4* p = (const float4*)(rowbase + k0);
  const float4* q = (const float4*)(rowbase + k0 + 16);
  float4 a0 = p[0], a1 = p[1], b0 = q[0], b1 = q[1];
  FragBF f;
  f.u[0] = pk2(a0.x, a0.y); f.u[1] = pk2(a0.z, a0.w);
  f.u[2] = pk2(a1.x, a1.y); f.u[3] = pk2(a1.z, a1.w);
  f.u[4] = pk2(b0.x, b0.y); f.u[5] = pk2(b0.z, b0.w);
  f.u[6] = pk2(b1.x, b1.y); f.u[7] = pk2(b1.z, b1.w);
  return f.v;
}

#define WMMA_BF16(A, B, C) \
  __builtin_amdgcn_wmma_f32_16x16x32_bf16(false, (A), false, (B), (short)0, (C), false, false)

// One 16-row x 16-col output tile, A rows in LDS (bf16), B rows in global (f32).
template <int KC>
__device__ __forceinline__ v8f mm_row(const unsigned short* abase, int astride,
                                      const float* brow, int lane) {
  v8f c = {0.f, 0.f, 0.f, 0.f, 0.f, 0.f, 0.f, 0.f};
  const unsigned short* arow = abase + (lane & 15) * astride;
#pragma unroll
  for (int kc = 0; kc < KC; ++kc) {
    v16bf af = load_frag_lds(arow, kc, lane);
    v16bf bf = load_frag_gf32(brow, kc, lane);
    c = WMMA_BF16(af, bf, c);
  }
  return c;
}

// ---------------------------------------------------------------------------
// Kernel A: fused GNN torso. One workgroup per graph; x (512x128) resident in
// LDS as bf16 for the whole 3-layer GNN + pooling. One WMMA matmul per layer
// via the algebraic fusion  h = x @ (Wr - cg*Wl)^T + (bl + cg * s@Wl^T).
// ---------------------------------------------------------------------------
#define A_OFF_XBF   0                         // 512*128 bf16 = 131072
#define A_OFF_BEFF  131072                    // 128*128 bf16 = 32768 (also raw frame0 stage)
#define A_OFF_MASK  (131072 + 32768)          // 512 f32 = 2048
#define A_OFF_SVEC  (A_OFF_MASK + 2048)       // 128 f32
#define A_OFF_BIAS2 (A_OFF_SVEC + 512)        // 128 f32
#define A_OFF_RED   (A_OFF_BIAS2 + 512)       // 256 f32
#define A_OFF_NSH   (A_OFF_RED + 1024)        // 16
#define A_LDS_BYTES (A_OFF_NSH + 16)

__global__ __launch_bounds__(BDIM) void gnn_kernel(
    const int* __restrict__ xx, const float* __restrict__ ss,
    const float* __restrict__ Win, const float* __restrict__ b_in,
    const float* __restrict__ Wl, const float* __restrict__ bl,
    const float* __restrict__ Wr, const float* __restrict__ ln_g,
    const float* __restrict__ ln_b, float* __restrict__ out) {
  extern __shared__ char smem[];
  unsigned short* xbf   = (unsigned short*)(smem + A_OFF_XBF);
  unsigned short* beffT = (unsigned short*)(smem + A_OFF_BEFF);
  float* maskA = (float*)(smem + A_OFF_MASK);
  float* svec  = (float*)(smem + A_OFF_SVEC);
  float* bias2 = (float*)(smem + A_OFF_BIAS2);
  float* red   = (float*)(smem + A_OFF_RED);
  float* nsh   = (float*)(smem + A_OFF_NSH);

  const int b = blockIdx.x;
  const int tid = threadIdx.x;
  const int lane = tid & 31;
  const int wave = tid >> 5;
  const int row16 = lane & 15;
  const int half8 = (lane >> 4) << 3;

  // ---- async-stage frame0 (512 ints) into the (currently dead) Beff panel ----
  const int* rawA = (const int*)(smem + A_OFF_BEFF);
  {
    unsigned base = (unsigned)(size_t)rawA;
    if (tid < 128)
      async_copy16(base + tid * 16, xx + (size_t)b * TT * NN + tid * 4);
  }
  if (tid == 0) nsh[0] = 0.0f;
  wait_async0();
  __syncthreads();

  // ---- phase 0: node features -> x0 (lin_in), mask, active count ----
  const float sv = ss[b] * (1.0f / 8.0f);
  float cnt = 0.0f;
  for (int node = tid; node < NN; node += BDIM) {
    int f = rawA[node];
    float ff = (float)f;
    float mk = (f != 0) ? 1.0f : 0.0f;
    maskA[node] = mk;
    cnt += mk;
    float ci = (float)(node >> 6) * (1.0f / 7.0f);
    float cj = (float)((node >> 3) & 7) * (1.0f / 7.0f);
    float ck = (float)(node & 7) * (1.0f / 7.0f);
    float f3 = ff * 0.5f;
    unsigned short* xr = xbf + node * CD;
    for (int c = 0; c < CD; ++c) {
      const float* w = Win + c * 5;  // uniform across lanes -> s_load
      float v = w[0] * ci + w[1] * cj + w[2] * ck + w[3] * f3 + w[4] * sv + b_in[c];
      xr[c] = f2bf(v);
    }
  }
  atomicAdd(nsh, cnt);
  __syncthreads();
  const float nact = nsh[0];
  const float cg = (nact > 1.0f) ? 1.0f / (nact - 1.0f) : 0.0f;

  // per-lane LN params for this lane's output columns
  float gcol[8], bcol[8];
#pragma unroll
  for (int nt = 0; nt < 8; ++nt) {
    gcol[nt] = ln_g[nt * 16 + row16];
    bcol[nt] = ln_b[nt * 16 + row16];
  }

  for (int l = 0; l < LGNN; ++l) {
    const float* Wl_l = Wl + (size_t)l * CD * CD;
    const float* Wr_l = Wr + (size_t)l * CD * CD;
    if (lane == 0) {
      __builtin_prefetch(Wl_l, 0, 1);
      __builtin_prefetch(Wr_l, 0, 1);
    }
    // ---- masked column sums s[c] ----
    {
      int c = tid & 127, half = tid >> 7;
      float acc = 0.0f;
      int n0 = half * 256;
      for (int node = n0; node < n0 + 256; ++node)
        acc += bf2f(xbf[node * CD + c]) * maskA[node];
      red[tid] = acc;
    }
    __syncthreads();
    if (tid < CD) svec[tid] = red[tid] + red[tid + CD];
    __syncthreads();
    // ---- fused bias:  bias2[j] = bl[j] + cg * dot(s, Wl[j,:]) ----
    if (tid < CD) {
      const float* wr = Wl_l + tid * CD;
      float t = 0.0f;
      for (int k = 0; k < CD; ++k) t += svec[k] * wr[k];
      bias2[tid] = bl[l * CD + tid] + cg * t;
    }
    // ---- per-graph effective weight (N-major rows, K contiguous) ----
    for (int idx = tid; idx < CD * CD; idx += BDIM)
      beffT[idx] = f2bf(Wr_l[idx] - cg * Wl_l[idx]);
    __syncthreads();

    float b2col[8];
#pragma unroll
    for (int nt = 0; nt < 8; ++nt) b2col[nt] = bias2[nt * 16 + row16];

    // ---- h = x @ Beff + bias2; relu; layernorm; write back (in place) ----
    for (int mt = wave * 4; mt < wave * 4 + 4; ++mt) {
      const unsigned short* arow = xbf + (mt * 16 + row16) * CD;
      v16bf afr[4];
#pragma unroll
      for (int kc = 0; kc < 4; ++kc) afr[kc] = load_frag_lds(arow, kc, lane);

      v8f accs[8];
#pragma unroll
      for (int nt = 0; nt < 8; ++nt) {
        const unsigned short* brow = beffT + (nt * 16 + row16) * CD;
        v8f c = {0.f, 0.f, 0.f, 0.f, 0.f, 0.f, 0.f, 0.f};
#pragma unroll
        for (int kc = 0; kc < 4; ++kc) {
          v16bf bfr = load_frag_lds(brow, kc, lane);
          c = WMMA_BF16(afr[kc], bfr, c);
        }
        accs[nt] = c;
      }
      // relu + row stats (D rows split lanes 0-15 / 16-31 -> width-16 reduce)
      float mu[8], rs[8];
#pragma unroll
      for (int r = 0; r < 8; ++r) {
        float sm = 0.0f, sq = 0.0f;
#pragma unroll
        for (int nt = 0; nt < 8; ++nt) {
          float h = accs[nt][r] + b2col[nt];
          h = fmaxf(h, 0.0f);
          accs[nt][r] = h;
          sm += h; sq += h * h;
        }
#pragma unroll
        for (int m = 1; m < 16; m <<= 1) {
          sm += __shfl_xor(sm, m, 16);
          sq += __shfl_xor(sq, m, 16);
        }
        float m1 = sm * (1.0f / 128.0f);
        float var = sq * (1.0f / 128.0f) - m1 * m1;
        mu[r] = m1;
        rs[r] = rsqrtf(var + 1e-5f);
      }
#pragma unroll
      for (int r = 0; r < 8; ++r) {
        unsigned short* orow = xbf + (mt * 16 + r + half8) * CD;
#pragma unroll
        for (int nt = 0; nt < 8; ++nt) {
          float y = (accs[nt][r] - mu[r]) * rs[r] * gcol[nt] + bcol[nt];
          orow[nt * 16 + row16] = f2bf(y);
        }
      }
    }
    __syncthreads();
  }

  // ---- axis-slice mean pooling of masked x -> out rows 0..23 ----
  for (int idx = tid; idx < 24 * CD; idx += BDIM) {
    int p = idx >> 7, c = idx & 127;
    float acc = 0.0f;
    if (p < 8) {
      int base = p << 6;
      for (int q = 0; q < 64; ++q) {
        int node = base + q;
        acc += bf2f(xbf[node * CD + c]) * maskA[node];
      }
    } else if (p < 16) {
      int j = p - 8;
      for (int q = 0; q < 64; ++q) {
        int node = ((q >> 3) << 6) + (j << 3) + (q & 7);
        acc += bf2f(xbf[node * CD + c]) * maskA[node];
      }
    } else {
      int k = p - 16;
      for (int q = 0; q < 64; ++q) {
        int node = ((q >> 3) << 6) + ((q & 7) << 3) + k;
        acc += bf2f(xbf[node * CD + c]) * maskA[node];
      }
    }
    out[(size_t)b * 26 * CD + p * CD + c] = acc * (1.0f / 64.0f);
  }
}

// ---------------------------------------------------------------------------
// Kernel B: transformer over 15 action tokens (+ mv head). One WG per batch.
// Sequence padded to 16 rows; all projections use WMMA with B-fragments read
// directly from global fp32 weights. Raw action frames are async-staged into
// the (dead at that point) aL/qkv LDS region, then converted to bf16 once.
// ---------------------------------------------------------------------------
#define B_OFF_A    0                        // 16*128 f32 = 8192
#define B_OFF_QKV  8192                     // 16*384 f32 = 24576
#define B_OFF_HBF  (8192 + 24576)           // 16*128 bf16 = 4096
#define B_OFF_OBF  (B_OFF_HBF + 4096)       // 16*128 bf16 = 4096
#define B_OFF_BIG  (B_OFF_OBF + 4096)       // 16*512 bf16 = 16384 (frames / ffn)
#define B_OFF_REDM (B_OFF_BIG + 16384)      // 256 f32
#define B_OFF_REDV (B_OFF_REDM + 1024)      // 256 f32
#define B_OFF_MU   (B_OFF_REDV + 1024)      // 16 f32
#define B_OFF_RS   (B_OFF_MU + 64)          // 16 f32
#define B_LDS_BYTES (B_OFF_RS + 64)

__device__ void ln_to_bf(const float* aL, unsigned short* dst,
                         const float* g, const float* be,
                         float* redm, float* redv, float* muv, float* rsv) {
  const int tid = threadIdx.x;
  {
    int token = tid >> 4, part = tid & 15;
    const float* row = aL + token * CD + part * 8;
    float sm = 0.0f, sq = 0.0f;
#pragma unroll
    for (int i = 0; i < 8; ++i) { float x = row[i]; sm += x; sq += x * x; }
    redm[tid] = sm; redv[tid] = sq;
  }
  __syncthreads();
  if (tid < 16) {
    float sm = 0.0f, sq = 0.0f;
    for (int i = 0; i < 16; ++i) { sm += redm[tid * 16 + i]; sq += redv[tid * 16 + i]; }
    float mu = sm * (1.0f / 128.0f);
    float var = sq * (1.0f / 128.0f) - mu * mu;
    muv[tid] = mu;
    rsv[tid] = rsqrtf(var + 1e-5f);
  }
  __syncthreads();
  for (int idx = tid; idx < 16 * CD; idx += BDIM) {
    int token = idx >> 7, c = idx & 127;
    dst[idx] = f2bf((aL[idx] - muv[token]) * rsv[token] * g[c] + be[c]);
  }
  __syncthreads();
}

__global__ __launch_bounds__(BDIM) void tf_kernel(
    const int* __restrict__ xx, const float* __restrict__ ss,
    const float* __restrict__ Wqkv, const float* __restrict__ bqkv,
    const float* __restrict__ Wo, const float* __restrict__ bo,
    const float* __restrict__ ln1g, const float* __restrict__ ln1b,
    const float* __restrict__ ln2g, const float* __restrict__ ln2b,
    const float* __restrict__ W1, const float* __restrict__ b1,
    const float* __restrict__ W2, const float* __restrict__ b2,
    const float* __restrict__ Wact, const float* __restrict__ bact,
    const float* __restrict__ Wsc, const float* __restrict__ bsc,
    float* __restrict__ out) {
  extern __shared__ char smem[];
  float* aL            = (float*)(smem + B_OFF_A);
  float* qkv           = (float*)(smem + B_OFF_QKV);
  unsigned short* hbf  = (unsigned short*)(smem + B_OFF_HBF);
  unsigned short* obf  = (unsigned short*)(smem + B_OFF_OBF);
  unsigned short* big  = (unsigned short*)(smem + B_OFF_BIG);
  float* redm = (float*)(smem + B_OFF_REDM);
  float* redv = (float*)(smem + B_OFF_REDV);
  float* muv  = (float*)(smem + B_OFF_MU);
  float* rsv  = (float*)(smem + B_OFF_RS);

  const int b = blockIdx.x;
  const int tid = threadIdx.x;
  const int lane = tid & 31;
  const int wave = tid >> 5;
  const int row16 = lane & 15;
  const int half8 = (lane >> 4) << 3;

  // ---- async-stage frames 1..15 (7680 ints = 30 KB) into aL/qkv region ----
  {
    unsigned base = (unsigned)(size_t)smem;
    const int* gsrc = xx + (size_t)b * TT * NN + NN;  // frame 1 onward
    for (int idx = tid; idx < (SEQ * NN) / 4; idx += BDIM)  // 1920 x 16B
      async_copy16(base + idx * 16, gsrc + idx * 4);
  }
  wait_async0();
  __syncthreads();

  // ---- convert staged ints -> bf16 token rows (token 15 = zero pad) ----
  {
    const int* rawB = (const int*)smem;
    for (int idx = tid; idx < 16 * NN; idx += BDIM) {
      int token = idx >> 9;
      unsigned short v = 0;
      if (token < SEQ) v = f2bf((float)rawB[idx]);
      big[idx] = v;
    }
  }
  __syncthreads();

  // ---- a = frames @ Wact^T + bact  (K=512, 8 N-tiles -> one per wave) ----
  {
    int nt = wave;
    const float* brow = Wact + (size_t)(nt * 16 + row16) * 512;
    v8f c = mm_row<16>(big, 512, brow, lane);
    int col = nt * 16 + row16;
    float bb = bact[col];
#pragma unroll
    for (int r = 0; r < 8; ++r) aL[(r + half8) * CD + col] = c[r] + bb;
  }
  __syncthreads();

  for (int l = 0; l < LTF; ++l) {
    const float* Wqkv_l = Wqkv + (size_t)l * 3 * CD * CD;
    const float* bqkv_l = bqkv + l * 3 * CD;
    const float* Wo_l = Wo + (size_t)l * CD * CD;
    const float* bo_l = bo + l * CD;
    const float* W1_l = W1 + (size_t)l * FFD * CD;
    const float* b1_l = b1 + l * FFD;
    const float* W2_l = W2 + (size_t)l * CD * FFD;
    const float* b2_l = b2 + l * CD;
    if (lane == 0) {
      __builtin_prefetch(Wqkv_l, 0, 1);
      __builtin_prefetch(W1_l, 0, 1);
    }

    ln_to_bf(aL, hbf, ln1g + l * CD, ln1b + l * CD, redm, redv, muv, rsv);

    // ---- QKV projection: 24 N-tiles, 3 per wave ----
#pragma unroll
    for (int i = 0; i < 3; ++i) {
      int nt = wave * 3 + i;
      const float* brow = Wqkv_l + (size_t)(nt * 16 + row16) * CD;
      v8f c = mm_row<4>(hbf, CD, brow, lane);
      int col = nt * 16 + row16;
      float bb = bqkv_l[col];
#pragma unroll
      for (int r = 0; r < 8; ++r) qkv[(r + half8) * 384 + col] = c[r] + bb;
    }
    __syncthreads();

    // ---- attention (4 heads x 15 queries, tiny -> VALU) ----
    if (tid < 4 * SEQ) {
      int h = tid / SEQ, q = tid - h * SEQ;
      const float* qp = qkv + q * 384 + h * 32;
      float sc[SEQ];
      float mx = -1e30f;
#pragma unroll
      for (int k2 = 0; k2 < SEQ; ++k2) {
        const float* kp = qkv + k2 * 384 + CD + h * 32;
        float d = 0.0f;
#pragma unroll
        for (int dd = 0; dd < 32; ++dd) d += qp[dd] * kp[dd];
        d *= 0.1767766952966369f;  // 1/sqrt(32)
        sc[k2] = d;
        mx = fmaxf(mx, d);
      }
      float sum = 0.0f;
#pragma unroll
      for (int k2 = 0; k2 < SEQ; ++k2) { float e = __expf(sc[k2] - mx); sc[k2] = e; sum += e; }
      float inv = 1.0f / sum;
#pragma unroll
      for (int dd = 0; dd < 32; ++dd) {
        float o = 0.0f;
#pragma unroll
        for (int k2 = 0; k2 < SEQ; ++k2) o += sc[k2] * qkv[k2 * 384 + 256 + h * 32 + dd];
        obf[q * CD + h * 32 + dd] = f2bf(o * inv);
      }
    }
    if (tid < CD) obf[SEQ * CD + tid] = 0;  // zero pad row
    __syncthreads();

    // ---- output projection + residual ----
    {
      int nt = wave;
      const float* brow = Wo_l + (size_t)(nt * 16 + row16) * CD;
      v8f c = mm_row<4>(obf, CD, brow, lane);
      int col = nt * 16 + row16;
      float bb = bo_l[col];
#pragma unroll
      for (int r = 0; r < 8; ++r) aL[(r + half8) * CD + col] += c[r] + bb;
    }
    __syncthreads();

    ln_to_bf(aL, hbf, ln2g + l * CD, ln2b + l * CD, redm, redv, muv, rsv);

    // ---- FFN up + relu: 32 N-tiles, 4 per wave ----
#pragma unroll
    for (int i = 0; i < 4; ++i) {
      int nt = wave * 4 + i;
      const float* brow = W1_l + (size_t)(nt * 16 + row16) * CD;
      v8f c = mm_row<4>(hbf, CD, brow, lane);
      int col = nt * 16 + row16;
      float bb = b1_l[col];
#pragma unroll
      for (int r = 0; r < 8; ++r)
        big[(r + half8) * FFD + col] = f2bf(fmaxf(c[r] + bb, 0.0f));
    }
    __syncthreads();

    // ---- FFN down + residual (K=512) ----
    {
      int nt = wave;
      const float* brow = W2_l + (size_t)(nt * 16 + row16) * FFD;
      v8f c = mm_row<16>(big, FFD, brow, lane);
      int col = nt * 16 + row16;
      float bb = b2_l[col];
#pragma unroll
      for (int r = 0; r < 8; ++r) aL[(r + half8) * CD + col] += c[r] + bb;
    }
    __syncthreads();
  }

  // ---- act_emb (mean over 15 tokens) and mv_emb heads ----
  if (tid < CD) {
    float s = 0.0f;
    for (int t = 0; t < SEQ; ++t) s += aL[t * CD + tid];
    out[(size_t)b * 26 * CD + 24 * CD + tid] = s * (1.0f / 15.0f);
    float mv = ss[b] * Wsc[tid] + bsc[tid];
    out[(size_t)b * 26 * CD + 25 * CD + tid] = fmaxf(mv, 0.0f);
  }
}

extern "C" void kernel_launch(void* const* d_in, const int* in_sizes, int n_in,
                              void* d_out, int out_size, void* d_ws, size_t ws_size,
                              hipStream_t stream) {
  (void)in_sizes; (void)n_in; (void)out_size; (void)d_ws; (void)ws_size;
  const int*   xx   = (const int*)  d_in[0];
  const float* ss_  = (const float*)d_in[1];
  const float* Win  = (const float*)d_in[2];
  const float* b_in = (const float*)d_in[3];
  const float* Wl   = (const float*)d_in[4];
  const float* bl   = (const float*)d_in[5];
  const float* Wr   = (const float*)d_in[6];
  const float* ln_g = (const float*)d_in[7];
  const float* ln_b = (const float*)d_in[8];
  const float* Wqkv = (const float*)d_in[9];
  const float* bqkv = (const float*)d_in[10];
  const float* Wo   = (const float*)d_in[11];
  const float* bo   = (const float*)d_in[12];
  const float* ln1g = (const float*)d_in[13];
  const float* ln1b = (const float*)d_in[14];
  const float* ln2g = (const float*)d_in[15];
  const float* ln2b = (const float*)d_in[16];
  const float* W1   = (const float*)d_in[17];
  const float* b1   = (const float*)d_in[18];
  const float* W2   = (const float*)d_in[19];
  const float* b2   = (const float*)d_in[20];
  const float* Wact = (const float*)d_in[21];
  const float* bact = (const float*)d_in[22];
  const float* Wsc  = (const float*)d_in[23];
  const float* bsc  = (const float*)d_in[24];
  float* out = (float*)d_out;

  gnn_kernel<<<dim3(NB), dim3(BDIM), A_LDS_BYTES, stream>>>(
      xx, ss_, Win, b_in, Wl, bl, Wr, ln_g, ln_b, out);
  tf_kernel<<<dim3(NB), dim3(BDIM), B_LDS_BYTES, stream>>>(
      xx, ss_, Wqkv, bqkv, Wo, bo, ln1g, ln1b, ln2g, ln2b,
      W1, b1, W2, b2, Wact, bact, Wsc, bsc, out);
}
